// Net_89094801588965
// MI455X (gfx1250) — matrix operand order repaired and checked by reading, not compiled, not run on hardware
//
#include <hip/hip_runtime.h>
#include <cstdint>
#include <cstddef>

// ---------------- problem constants ----------------
constexpr int BN   = 65536;
constexpr int DIN  = 784;
constexpr int HID  = 400;
constexpr int LAT  = 256;
constexpr int NT   = 10;
constexpr int H1N  = 28;
constexpr int H2N  = 28;
constexpr int NC   = 10;
constexpr int X2   = 2 * LAT;     // 512
constexpr int NH1  = NT * H1N;    // 280 (all-task a1 width)

// ---------------- WMMA types ----------------
typedef __attribute__((ext_vector_type(16))) __bf16 v16bf;
typedef __attribute__((ext_vector_type(8)))  float  v8f;

union Frag {
    v16bf    v;
    unsigned u[8];
    uint4    q[2];
};

__device__ __forceinline__ unsigned short f2bf(float f) {
    union { float f; unsigned u; } x; x.f = f;
    unsigned u = x.u;
    return (unsigned short)((u + 0x7FFFu + ((u >> 16) & 1u)) >> 16);  // RNE
}
__device__ __forceinline__ float bf2f(unsigned short b) {
    union { unsigned u; float f; } x; x.u = ((unsigned)b) << 16;
    return x.f;
}
__device__ __forceinline__ unsigned pack2(unsigned short lo, unsigned short hi) {
    return (unsigned)lo | ((unsigned)hi << 16);
}

// ---------------------------------------------------------------------------
// Weight converter: f32 row-major [K,N] (optionally task-blocked along columns)
// -> bf16 B-fragment layout: frag[((kt*ntPad + nt)*32 + lane)*8 + v]
//   packed pair (k, k+1) where k = kt*32 + (lane>>4)*8 + 2(v&3) + 16(v>>2), n = nt*16 + (lane&15)
// Zero-pads k >= K and n >= N (including whole pad tiles nt*16 >= N), so the
// GEMM inner loop can run all 4 N-fragments unconditionally.
// ---------------------------------------------------------------------------
__global__ void convert_wfrag(const float* __restrict__ W, int K, int N,
                              int rowStride, int perTaskN, long colTaskPitch,
                              unsigned* __restrict__ out, int ktTot, int ntPad,
                              const int* __restrict__ devTaskBase, long taskBaseStride)
{
    int idx = blockIdx.x * blockDim.x + threadIdx.x;
    int total = ktTot * ntPad * 256;
    if (idx >= total) return;
    if (devTaskBase) W += (long)(*devTaskBase) * taskBaseStride;

    int v    = idx & 7;
    int lane = (idx >> 3) & 31;
    int tile = idx >> 8;
    int nt   = tile % ntPad;
    int kt   = tile / ntPad;

    int n  = nt * 16 + (lane & 15);
    int kb = kt * 32 + (lane >> 4) * 8 + (v & 3) * 2 + (v >> 2) * 16;

    unsigned short lo = 0, hi = 0;
    if (n < N) {
        int t = n / perTaskN, c = n % perTaskN;
        const float* p = W + (long)t * colTaskPitch + c;
        if (kb     < K) lo = f2bf(p[(long)kb       * rowStride]);
        if (kb + 1 < K) hi = f2bf(p[(long)(kb + 1) * rowStride]);
    }
    out[idx] = pack2(lo, hi);
}

// ---------------------------------------------------------------------------
// 4 back-to-back WMMAs against consecutive 16-wide N-tiles (branch-free).
// Wtile points at fragment tile (kt, ntBase); tiles are 256 dwords apart.
// ---------------------------------------------------------------------------
__device__ __forceinline__ void wmma4(const unsigned* __restrict__ Wtile, int lane,
                                      const v16bf& av, v8f acc[4])
{
#pragma unroll
    for (int j = 0; j < 4; ++j) {
        Frag bf;
        const uint4* wp = (const uint4*)(Wtile + j * 256 + lane * 8);
        bf.q[0] = wp[0];
        bf.q[1] = wp[1];
        acc[j] = __builtin_amdgcn_wmma_f32_16x16x32_bf16(
            false, av, false, bf.v, (short)0, acc[j], false, false);
    }
}

// ---------------------------------------------------------------------------
// bf16 WMMA GEMM:  C[B, N] = act(A[B, K] @ Wfrag + bias)
//   AT = float (converted to bf16 on the fly) or unsigned short (bf16 bits)
//   Workgroup: 256 threads = 8 waves; tile 128(M) x 64(N); 4 accumulators/wave.
//   Inner K-loop is branch-free (full tiles); the K-tail runs once, guarded.
//   blockIdx.z strides (zA/zW/zBias/zCol) implement per-task batched GEMMs.
// ---------------------------------------------------------------------------
template <typename AT, bool RELU>
__global__ __launch_bounds__(256)
void gemm_wmma_bf16(const AT* __restrict__ A, int lda, int K,
                    const unsigned* __restrict__ Wfrag, int ntPad,
                    const float* __restrict__ bias,
                    unsigned short* __restrict__ Cout, int ldc, int col0, int N,
                    const int* __restrict__ devTaskBias, int taskBiasStride,
                    int zA, int zW, int zBias, int zCol)
{
    const int tz = blockIdx.z;
    A     += (size_t)tz * (size_t)zA;      // column shift within each row
    Wfrag += (size_t)tz * (size_t)zW;
    const float* bptr = bias + (size_t)tz * (size_t)zBias;
    if (devTaskBias) bptr += (size_t)(*devTaskBias) * (size_t)taskBiasStride;
    const int c0 = col0 + tz * zCol;

    const int lane    = threadIdx.x & 31;
    const int wave    = threadIdx.x >> 5;
    const int rowBase = blockIdx.x * 128 + wave * 16;
    const int ntBase  = blockIdx.y * 4;

    v8f acc[4];
#pragma unroll
    for (int j = 0; j < 4; ++j) acc[j] = (v8f){};

    const int hsel = lane >> 4;
    const int m    = lane & 15;
    const AT* Arow = A + (size_t)(rowBase + m) * (size_t)lda;
    const unsigned* Wrow = Wfrag + (size_t)ntBase * 256;   // tile (0, ntBase)
    const size_t    Wstride = (size_t)ntPad * 256;         // dwords per kt row

    const int ktFull = K >> 5;
    for (int kt = 0; kt < ktFull; ++kt) {
        const int kb = kt * 32 + hsel * 8;
        Frag af;
        if constexpr (sizeof(AT) == 4) {          // f32 source, convert on the fly
#pragma unroll
            for (int g = 0; g < 2; ++g) {
                const float4* p = (const float4*)((const float*)Arow + kb + g * 16);
                float4 f0 = p[0], f1 = p[1];
                af.u[g * 4 + 0] = pack2(f2bf(f0.x), f2bf(f0.y));
                af.u[g * 4 + 1] = pack2(f2bf(f0.z), f2bf(f0.w));
                af.u[g * 4 + 2] = pack2(f2bf(f1.x), f2bf(f1.y));
                af.u[g * 4 + 3] = pack2(f2bf(f1.z), f2bf(f1.w));
            }
        } else {                                   // bf16 source: two b128 loads
#pragma unroll
            for (int g = 0; g < 2; ++g) {
                const uint4* p = (const uint4*)((const unsigned short*)Arow + kb + g * 16);
                af.q[g] = p[0];
            }
        }
        wmma4(Wrow + (size_t)kt * Wstride, lane, af.v, acc);
    }

    if (K & 31) {                                  // single guarded K-tail step
        const int kt = ktFull;
        const int kb = kt * 32 + hsel * 8;
        Frag af;
#pragma unroll
        for (int g = 0; g < 2; ++g)
#pragma unroll
            for (int vv = 0; vv < 4; ++vv) {
                int k0 = kb + g * 16 + 2 * vv;
                unsigned short lo = 0, hi = 0;
                if constexpr (sizeof(AT) == 4) {
                    if (k0     < K) lo = f2bf(((const float*)Arow)[k0]);
                    if (k0 + 1 < K) hi = f2bf(((const float*)Arow)[k0 + 1]);
                } else {
                    if (k0     < K) lo = ((const unsigned short*)Arow)[k0];
                    if (k0 + 1 < K) hi = ((const unsigned short*)Arow)[k0 + 1];
                }
                af.u[g * 4 + vv] = pack2(lo, hi);
            }
        wmma4(Wrow + (size_t)kt * Wstride, lane, af.v, acc);
    }

    // ---- epilogue: bias + optional ReLU, store bf16 (pad tiles fail c<N) ----
#pragma unroll
    for (int j = 0; j < 4; ++j) {
        int c = (ntBase + j) * 16 + m;
        if (c >= N) continue;
        float bv = bptr[c];
#pragma unroll
        for (int vv = 0; vv < 8; ++vv) {
            int   row = rowBase + hsel * 8 + vv;
            float val = acc[j][vv] + bv;
            if (RELU) val = fmaxf(val, 0.f);
            Cout[(size_t)row * ldc + c0 + c] = f2bf(val);
        }
    }
}

// ---------------------------------------------------------------------------
// Final stage: per-sample task select + tiny 28->10 linear (+bias), f32 out.
// One thread per (sample, class-lane<16). Weight reads are L2-resident.
// ---------------------------------------------------------------------------
__global__ void head3_select(const unsigned short* __restrict__ A2, // [B, 280] bf16
                             const int* __restrict__ tt,
                             const float* __restrict__ h3_w,        // [T,28,10]
                             const float* __restrict__ h3_b,        // [T,10]
                             float* __restrict__ out)               // [B,10]
{
    int gid = blockIdx.x * blockDim.x + threadIdx.x;
    int b = gid >> 4, c = gid & 15;
    if (b >= BN || c >= NC) return;
    int t = tt[b];
    const unsigned short* a = A2 + (size_t)b * NH1 + t * H2N;
    const float* w = h3_w + (size_t)t * H2N * NC + c;
    float s = h3_b[t * NC + c];
#pragma unroll
    for (int k = 0; k < H2N; ++k) s += bf2f(a[k]) * w[k * NC];
    out[(size_t)b * NC + c] = s;
}

// ---------------------------------------------------------------------------
// Host-side orchestration
// ---------------------------------------------------------------------------
static inline unsigned* U32(void* ws, size_t byteOff) {
    return (unsigned*)((char*)ws + byteOff);
}
static inline unsigned short* U16(void* ws, size_t byteOff) {
    return (unsigned short*)((char*)ws + byteOff);
}

extern "C" void kernel_launch(void* const* d_in, const int* in_sizes, int n_in,
                              void* d_out, int out_size, void* d_ws, size_t ws_size,
                              hipStream_t stream) {
    const float* x_s    = (const float*)d_in[0];
    const int*   tt     = (const int*)d_in[2];
    const int*   taskId = (const int*)d_in[3];
    const float* fc1_w  = (const float*)d_in[4];
    const float* fc1_b  = (const float*)d_in[5];
    const float* fc2_w  = (const float*)d_in[6];
    const float* fc2_b  = (const float*)d_in[7];
    const float* fc3_w  = (const float*)d_in[8];
    const float* fc3_b  = (const float*)d_in[9];
    const float* priv_w = (const float*)d_in[10];
    const float* priv_b = (const float*)d_in[11];
    const float* h1_w   = (const float*)d_in[12];
    const float* h1_b   = (const float*)d_in[13];  // [T,28] flat == [280]
    const float* h2_w   = (const float*)d_in[14];
    const float* h2_b   = (const float*)d_in[15];
    const float* h3_w   = (const float*)d_in[16];
    const float* h3_b   = (const float*)d_in[17];
    float* out = (float*)d_out;

    // fragment tile counts: K padded to 32; N padded to 16 then to 4-tile groups
    constexpr int ktFC1 = 25, npFC1 = 28;   // 784x400  (25 tiles -> pad 28)
    constexpr int ktFC2 = 13, npFC2 = 28;   // 400x400
    constexpr int ktFC3 = 13, npFC3 = 16;   // 400x256
    constexpr int ktPRV = 25, npPRV = 16;   // 784x256
    constexpr int ktW1  = 16, npW1  = 20;   // 512x280  (18 -> 20)
    constexpr int ktW2  = 1,  npW2  = 4;    // 28x28    (2 -> 4), per task
    constexpr int W2FRAG = ktW2 * npW2 * 256;  // 1024 dwords per task

    // workspace layout (bytes, 256-aligned)
    const size_t oFC1F = 0;
    const size_t oFC2F = oFC1F + (size_t)ktFC1 * npFC1 * 256 * 4;  // +716800
    const size_t oFC3F = oFC2F + (size_t)ktFC2 * npFC2 * 256 * 4;  // +372736
    const size_t oPRVF = oFC3F + (size_t)ktFC3 * npFC3 * 256 * 4;  // +212992
    const size_t oW1F  = oPRVF + (size_t)ktPRV * npPRV * 256 * 4;  // +409600
    const size_t oW2F  = oW1F  + (size_t)ktW1  * npW1  * 256 * 4;  // +327680
    const size_t oH1   = oW2F  + (size_t)NT * W2FRAG * 4;          // +40960
    const size_t oH2   = oH1 + (size_t)BN * HID * 2;               // 52.4 MB (reused as a1)
    const size_t oCC   = oH2 + (size_t)BN * HID * 2;               // 52.4 MB (reused as a2)
    // oCC..oCC + B*512*2  (67.1 MB concat buffer); total ~174 MB

    unsigned *fc1f = U32(d_ws, oFC1F), *fc2f = U32(d_ws, oFC2F), *fc3f = U32(d_ws, oFC3F);
    unsigned *prvf = U32(d_ws, oPRVF), *w1f = U32(d_ws, oW1F),  *w2f  = U32(d_ws, oW2F);
    unsigned short *H1 = U16(d_ws, oH1), *H2 = U16(d_ws, oH2), *CC = U16(d_ws, oCC);
    unsigned short *A1 = H1, *A2 = H2;   // safe reuse: last reads precede these writes

    auto cblk = [](int dwords) { return (dwords + 255) / 256; };

    // 1) weight conversion to bf16 B-fragment layout (zero-padded K and N)
    convert_wfrag<<<cblk(ktFC1 * npFC1 * 256), 256, 0, stream>>>(
        fc1_w, DIN, HID, HID, HID, 0, fc1f, ktFC1, npFC1, nullptr, 0);
    convert_wfrag<<<cblk(ktFC2 * npFC2 * 256), 256, 0, stream>>>(
        fc2_w, HID, HID, HID, HID, 0, fc2f, ktFC2, npFC2, nullptr, 0);
    convert_wfrag<<<cblk(ktFC3 * npFC3 * 256), 256, 0, stream>>>(
        fc3_w, HID, LAT, LAT, LAT, 0, fc3f, ktFC3, npFC3, nullptr, 0);
    convert_wfrag<<<cblk(ktPRV * npPRV * 256), 256, 0, stream>>>(
        priv_w, DIN, LAT, LAT, LAT, 0, prvf, ktPRV, npPRV,
        taskId, (long)DIN * LAT);                                  // priv_w[task]
    convert_wfrag<<<cblk(ktW1 * npW1 * 256), 256, 0, stream>>>(
        h1_w, X2, NH1, H1N, H1N, (long)X2 * H1N, w1f, ktW1, npW1,  // all-task gather
        nullptr, 0);
    for (int t = 0; t < NT; ++t)
        convert_wfrag<<<cblk(W2FRAG), 256, 0, stream>>>(
            h2_w + (size_t)t * H1N * H2N, H1N, H2N, H2N, H2N, 0,
            w2f + (size_t)t * W2FRAG, ktW2, npW2, nullptr, 0);

    const dim3 blk(256);
    const int gx = BN / 128;  // 512

    // 2) shared MLP
    gemm_wmma_bf16<float, true><<<dim3(gx, npFC1 / 4), blk, 0, stream>>>(
        x_s, DIN, DIN, fc1f, npFC1, fc1_b, H1, HID, 0, HID, nullptr, 0, 0, 0, 0, 0);
    gemm_wmma_bf16<unsigned short, true><<<dim3(gx, npFC2 / 4), blk, 0, stream>>>(
        H1, HID, HID, fc2f, npFC2, fc2_b, H2, HID, 0, HID, nullptr, 0, 0, 0, 0, 0);
    gemm_wmma_bf16<unsigned short, true><<<dim3(gx, npFC3 / 4), blk, 0, stream>>>(
        H2, HID, HID, fc3f, npFC3, fc3_b, CC, X2, LAT, LAT, nullptr, 0, 0, 0, 0, 0);

    // 3) private branch (bias row chosen by device-side task_id) -> concat cols [0,256)
    gemm_wmma_bf16<float, true><<<dim3(gx, npPRV / 4), blk, 0, stream>>>(
        x_s, DIN, DIN, prvf, npPRV, priv_b, CC, X2, 0, LAT, taskId, LAT, 0, 0, 0, 0);

    // 4) all-task head layer 1: [B,512] @ [512,280]
    gemm_wmma_bf16<unsigned short, true><<<dim3(gx, npW1 / 4), blk, 0, stream>>>(
        CC, X2, X2, w1f, npW1, h1_b, A1, NH1, 0, NH1, nullptr, 0, 0, 0, 0, 0);

    // 5) per-task head layer 2 (z = task): K=28 tail-guarded so no column bleed
    gemm_wmma_bf16<unsigned short, true><<<dim3(gx, npW2 / 4, NT), blk, 0, stream>>>(
        A1, NH1, H1N, w2f, npW2, h2_b, A2, NH1, 0, H2N,
        nullptr, 0, /*zA*/ H1N, /*zW*/ W2FRAG, /*zBias*/ H2N, /*zCol*/ H2N);

    // 6) select task + final 28->10 linear, f32 output
    head3_select<<<(BN * 16) / 256, blk, 0, stream>>>(A2, tt, h3_w, h3_b, out);
}
